// CrossAttention_19404662243472
// MI455X (gfx1250) — matrix operand order repaired and checked by reading, not compiled, not run on hardware
//
#include <hip/hip_runtime.h>
#include <hip/hip_bf16.h>
#include <stdint.h>

// ---------------------------------------------------------------------------
// CDNA5 (gfx1250) CrossAttention, f16 WMMA + f32 accumulate, flash attention
// with async global->LDS double-buffered K/V staging.
// Shapes (fixed by reference): B=2, dim=96, dim_c=192, hf=wf=128, ha=wa=64,
// heads=4, dim_head=64, inner=256, N=ha*wa=4096, NF=hf*wf=16384.
// ---------------------------------------------------------------------------

typedef __attribute__((ext_vector_type(16))) _Float16 v16h;
typedef __attribute__((ext_vector_type(8)))  _Float16 h8;
typedef __attribute__((ext_vector_type(8)))  float    v8f;

union AF { v16h v; h8 q[2]; _Float16 h[16]; };

#define WMMA_F16(A, B, C) \
  __builtin_amdgcn_wmma_f32_16x16x32_f16(false, (A), false, (B), (short)0, (C), false, false)

static __device__ __forceinline__ int imin(int a, int b) { return a < b ? a : b; }

// async copy of 16 bytes per lane: global (64-bit addr) -> LDS (32-bit offset)
static __device__ __forceinline__ void async_b128(unsigned lds_off, const void* gptr) {
  asm volatile("global_load_async_to_lds_b128 %0, %1, off"
               :: "v"(lds_off), "v"((unsigned long long)(uintptr_t)gptr)
               : "memory");
}

// ---------------------------------------------------------------------------
// f32 -> f16 weight conversion
// ---------------------------------------------------------------------------
__global__ void k_f32_to_f16(const float* __restrict__ in, _Float16* __restrict__ out, int n) {
  int i = blockIdx.x * blockDim.x + threadIdx.x;
  if (i < n) out[i] = (_Float16)in[i];
}

// ---------------------------------------------------------------------------
// PE + depthwise 3x3 on xc (192ch @64x64), three filters share the taps.
// Output pixel-major f16: y[b][p][c]  (GEMM B-operand friendly)
// ---------------------------------------------------------------------------
__global__ __launch_bounds__(192) void k_dw_xc(
    const float* __restrict__ x, const float* __restrict__ pw, const float* __restrict__ pb,
    const float* __restrict__ f0, const float* __restrict__ f1, const float* __restrict__ f2,
    _Float16* __restrict__ y0, _Float16* __restrict__ y1, _Float16* __restrict__ y2) {
  const int C = 192, H = 64, W = 64;
  int c = threadIdx.x, p = blockIdx.x, b = blockIdx.y;
  int hy = p / W, wx = p % W;
  float a0 = pw[c * 2], a1 = pw[c * 2 + 1], ab = pb[c];
  const float* xb = x + ((size_t)b * C + c) * (H * W);
  float s0 = 0.f, s1 = 0.f, s2 = 0.f;
#pragma unroll
  for (int ky = 0; ky < 3; ky++) {
    int yy = hy + ky - 1;
    if (yy < 0 || yy >= H) continue;
#pragma unroll
    for (int kx = 0; kx < 3; kx++) {
      int xx = wx + kx - 1;
      if (xx < 0 || xx >= W) continue;
      float v = xb[yy * W + xx] + a0 * (yy * (1.f / (H - 1))) + a1 * (xx * (1.f / (W - 1))) + ab;
      int ki = c * 9 + ky * 3 + kx;
      s0 += v * f0[ki]; s1 += v * f1[ki]; s2 += v * f2[ki];
    }
  }
  size_t o = ((size_t)b * (H * W) + p) * C + c;
  y0[o] = (_Float16)s0; y1[o] = (_Float16)s1; y2[o] = (_Float16)s2;
}

// PE + depthwise 3x3 on xf (96ch @128x128), one filter -> y[b][p][c] f16
__global__ __launch_bounds__(96) void k_dw_xf(
    const float* __restrict__ x, const float* __restrict__ pw, const float* __restrict__ pb,
    const float* __restrict__ f0, _Float16* __restrict__ y0) {
  const int C = 96, H = 128, W = 128;
  int c = threadIdx.x, p = blockIdx.x, b = blockIdx.y;
  int hy = p / W, wx = p % W;
  float a0 = pw[c * 2], a1 = pw[c * 2 + 1], ab = pb[c];
  const float* xb = x + ((size_t)b * C + c) * (H * W);
  float s0 = 0.f;
#pragma unroll
  for (int ky = 0; ky < 3; ky++) {
    int yy = hy + ky - 1;
    if (yy < 0 || yy >= H) continue;
#pragma unroll
    for (int kx = 0; kx < 3; kx++) {
      int xx = wx + kx - 1;
      if (xx < 0 || xx >= W) continue;
      float v = xb[yy * W + xx] + a0 * (yy * (1.f / (H - 1))) + a1 * (xx * (1.f / (W - 1))) + ab;
      s0 += v * f0[c * 9 + ky * 3 + kx];
    }
  }
  y0[((size_t)b * (H * W) + p) * C + c] = (_Float16)s0;
}

// depthwise 3x3 on attention output (256ch @64x64, f32 in), no PE -> y[b][p][c] f16
__global__ __launch_bounds__(256) void k_dw_o(
    const float* __restrict__ x, const float* __restrict__ f0, _Float16* __restrict__ y0) {
  const int C = 256, H = 64, W = 64;
  int c = threadIdx.x, p = blockIdx.x, b = blockIdx.y;
  int hy = p / W, wx = p % W;
  const float* xb = x + ((size_t)b * C + c) * (H * W);
  float s0 = 0.f;
#pragma unroll
  for (int ky = 0; ky < 3; ky++) {
    int yy = hy + ky - 1;
    if (yy < 0 || yy >= H) continue;
#pragma unroll
    for (int kx = 0; kx < 3; kx++) {
      int xx = wx + kx - 1;
      if (xx < 0 || xx >= W) continue;
      s0 += xb[yy * W + xx] * f0[c * 9 + ky * 3 + kx];
    }
  }
  y0[((size_t)b * (H * W) + p) * C + c] = (_Float16)s0;
}

// ---------------------------------------------------------------------------
// Pointwise conv as WMMA GEMM: out[b][o][p] = sum_c W[o][c] * Yt[b][p][c]
// One wave = one 16x16 (OxP) tile, K stepped by 32. All operand loads are
// contiguous 16B-per-lane, matching the gfx1250 WMMA A/B per-lane layouts.
// ---------------------------------------------------------------------------
__global__ __launch_bounds__(256) void k_gemm_pw(
    const _Float16* __restrict__ W,   // [O,C]
    const _Float16* __restrict__ Yt,  // [B,P,C]
    float* __restrict__ out,          // [B,O,P]
    int O, int C, int P) {
  int wave = threadIdx.x >> 5, lane = threadIdx.x & 31;
  int lm = lane & 15;
  int hoff = (lane >= 16) ? 8 : 0;
  int p0 = (blockIdx.x * 8 + wave) * 16;
  int o0 = blockIdx.y * 16;
  int b = blockIdx.z;
  const _Float16* Yb = Yt + ((size_t)b * P + p0 + lm) * C;
  const _Float16* Wr = W + (size_t)(o0 + lm) * C;
  v8f acc = {0.f, 0.f, 0.f, 0.f, 0.f, 0.f, 0.f, 0.f};
  for (int k = 0; k < C; k += 32) {
    AF a, bf;
    a.q[0] = *reinterpret_cast<const h8*>(Wr + k + hoff);        // K k..k+7 (lo) / +8..15 (hi)
    a.q[1] = *reinterpret_cast<const h8*>(Wr + k + 16 + hoff);   // K k+16.. / k+24..
    int kb = k + hoff;                                           // B: same K order as A
    bf.q[0] = *reinterpret_cast<const h8*>(Yb + kb);
    bf.q[1] = *reinterpret_cast<const h8*>(Yb + kb + 16);
    acc = WMMA_F16(a.v, bf.v, acc);
  }
  float* op = out + ((size_t)b * O + o0 + hoff) * P + p0 + lm;
#pragma unroll
  for (int r = 0; r < 8; r++) op[(size_t)r * P] = acc[r];
}

// ---------------------------------------------------------------------------
// Per-head LayerNorm over dim_head=64 with split_heads (ch = d*4 + head).
// In: x[b][256][4096] f32. Out: y[b*4+head][p][64] f16.  One wave per item.
// ---------------------------------------------------------------------------
__global__ __launch_bounds__(256) void k_ln_heads(
    const float* __restrict__ x, const float* __restrict__ g, const float* __restrict__ bt,
    _Float16* __restrict__ y) {
  int wave = threadIdx.x >> 5, lane = threadIdx.x & 31;
  int item = blockIdx.x * 8 + wave;  // 2*4*4096 = 32768
  int p = item & 4095;
  int bh = item >> 12;
  int head = bh & 3;
  int b = bh >> 2;
  const float* base = x + (size_t)b * 256 * 4096 + p;
  int d0 = lane, d1 = lane + 32;
  float v0 = base[(size_t)(d0 * 4 + head) * 4096];
  float v1 = base[(size_t)(d1 * 4 + head) * 4096];
  float s = v0 + v1;
  s += __shfl_xor(s, 1); s += __shfl_xor(s, 2); s += __shfl_xor(s, 4);
  s += __shfl_xor(s, 8); s += __shfl_xor(s, 16);
  float mu = s * (1.f / 64.f);
  float e0 = v0 - mu, e1 = v1 - mu;
  float qv = e0 * e0 + e1 * e1;
  qv += __shfl_xor(qv, 1); qv += __shfl_xor(qv, 2); qv += __shfl_xor(qv, 4);
  qv += __shfl_xor(qv, 8); qv += __shfl_xor(qv, 16);
  float rstd = rsqrtf(qv * (1.f / 64.f) + 1e-6f);
  _Float16* yo = y + ((size_t)bh * 4096 + p) * 64;
  yo[d0] = (_Float16)(e0 * rstd * g[head * 64 + d0] + bt[head * 64 + d0]);
  yo[d1] = (_Float16)(e1 * rstd * g[head * 64 + d1] + bt[head * 64 + d1]);
}

// ---------------------------------------------------------------------------
// V: bilinear 128->64 (align_corners) + split_heads + TRANSPOSE.
// Out: Vt[b*4+head][d][p(64x64)] f16  (PV B-operand: columns contiguous in p)
// ---------------------------------------------------------------------------
__global__ void k_v_resize(const float* __restrict__ vc, _Float16* __restrict__ vt) {
  int idx = blockIdx.x * blockDim.x + threadIdx.x;
  if (idx >= 2 * 4 * 64 * 4096) return;
  int p = idx & 4095;
  int d = (idx >> 12) & 63;
  int h = (idx >> 18) & 3;
  int b = idx >> 20;
  int oy = p >> 6, ox = p & 63;
  const float scl = 127.f / 63.f;
  float py = oy * scl, px = ox * scl;
  int y0 = imin((int)py, 126), x0 = imin((int)px, 126);
  float wy = py - y0, wx = px - x0;
  const float* src = vc + ((size_t)b * 256 + (d * 4 + h)) * 16384;
  int bs = y0 * 128 + x0;
  float v00 = src[bs], v01 = src[bs + 1], v10 = src[bs + 128], v11 = src[bs + 129];
  float v = (1.f - wy) * ((1.f - wx) * v00 + wx * v01) + wy * ((1.f - wx) * v10 + wx * v11);
  vt[idx] = (_Float16)v;
}

// ---------------------------------------------------------------------------
// Flash attention. One wave = 16 queries; the whole block (8 waves, same
// b/head) streams 4096 keys in chunks of 32. K (32x64) and V (64x32) chunks
// are staged once per block into LDS with double-buffered async copies
// (global_load_async_to_lds_b128, ASYNCcnt), overlapping the next chunk's
// copy with the current chunk's 8 WMMAs + streaming softmax.
// Qh/Kh: [bh][p][64] f16, Vt: [bh][64][p] f16. Out oc[b][d*4+head][p] f32.
// ---------------------------------------------------------------------------
__global__ __launch_bounds__(256) void k_attn(
    const _Float16* __restrict__ Qh, const _Float16* __restrict__ Kh,
    const _Float16* __restrict__ Vt, float* __restrict__ oc) {
  __shared__ alignas(16) _Float16 kbuf[2][32][64];  // [buf][key][d]
  __shared__ alignas(16) _Float16 vbuf[2][64][32];  // [buf][d][key]
  __shared__ alignas(16) _Float16 plds[8][16][40];  // per-wave P transpose tile

  const int tid = threadIdx.x;
  const int wave = tid >> 5, lane = tid & 31;
  const int lm = lane & 15;
  const bool hi = lane >= 16;
  const int hoff = hi ? 8 : 0;
  const int bh = blockIdx.y;
  const int head = bh & 3, b = bh >> 2;
  const int q0 = (blockIdx.x * 8 + wave) * 16;

  const _Float16* Qb = Qh + (size_t)bh * 4096 * 64;
  const _Float16* Kb = Kh + (size_t)bh * 4096 * 64;
  const _Float16* Vb = Vt + (size_t)bh * 64 * 4096;

  // per-thread 16B slices of the cooperative K/V chunk copies
  const int kd = tid * 8;            // halfs into the 32x64 K tile (4KB/chunk)
  const int vrow = tid >> 2;         // d row 0..63
  const int vseg = (tid & 3) * 8;    // halfs within 32-key row
  unsigned klds[2], vlds[2];
#pragma unroll
  for (int i = 0; i < 2; i++) {
    klds[i] = (unsigned)(uintptr_t)&kbuf[i][0][kd];   // low 32 bits = LDS offset
    vlds[i] = (unsigned)(uintptr_t)&vbuf[i][vrow][vseg];
  }

  AF qa0, qa1;  // A-fragments of Q, d-chunks 0..31 and 32..63
  {
    const _Float16* qr = Qb + (size_t)(q0 + lm) * 64;
    qa0.q[0] = *reinterpret_cast<const h8*>(qr + 0 + hoff);
    qa0.q[1] = *reinterpret_cast<const h8*>(qr + 16 + hoff);
    qa1.q[0] = *reinterpret_cast<const h8*>(qr + 32 + hoff);
    qa1.q[1] = *reinterpret_cast<const h8*>(qr + 48 + hoff);
  }

  float m[8], l[8], alpha[8];
  v8f o0 = {0.f,0.f,0.f,0.f,0.f,0.f,0.f,0.f};
  v8f o1 = o0, o2 = o0, o3 = o0;
#pragma unroll
  for (int r = 0; r < 8; r++) { m[r] = -3.0e38f; l[r] = 0.f; }
  const float sc = 0.125f;  // 64^-0.5

  // prologue: stage chunk 0 into buffer 0
  async_b128(klds[0], Kb + kd);
  async_b128(vlds[0], Vb + (size_t)vrow * 4096 + vseg);

  for (int c = 0; c < 128; c++) {
    const int j = c * 32;
    if (c + 1 < 128) {  // kick off next chunk into the other buffer
      const int jn = j + 32;
      async_b128(klds[(c + 1) & 1], Kb + (size_t)jn * 64 + kd);
      async_b128(vlds[(c + 1) & 1], Vb + (size_t)vrow * 4096 + jn + vseg);
      asm volatile("s_wait_asynccnt 0x2" ::: "memory");  // current chunk resident
    } else {
      asm volatile("s_wait_asynccnt 0x0" ::: "memory");
    }
    __syncthreads();  // publish cooperative copies to all waves

    const _Float16(*kk)[64] = kbuf[c & 1];
    const _Float16(*vv)[32] = vbuf[c & 1];

    // ---- S = scale * Q @ K^T, two 16x16 key-tiles ----
    AF kb;
    v8f s0 = {0.f,0.f,0.f,0.f,0.f,0.f,0.f,0.f};
    v8f s1 = s0;
    const _Float16* kr0 = &kk[lm][hoff];
    kb.q[0] = *reinterpret_cast<const h8*>(kr0);
    kb.q[1] = *reinterpret_cast<const h8*>(kr0 + 16);
    s0 = WMMA_F16(qa0.v, kb.v, s0);
    kb.q[0] = *reinterpret_cast<const h8*>(kr0 + 32);
    kb.q[1] = *reinterpret_cast<const h8*>(kr0 + 48);
    s0 = WMMA_F16(qa1.v, kb.v, s0);
    const _Float16* kr1 = &kk[16 + lm][hoff];
    kb.q[0] = *reinterpret_cast<const h8*>(kr1);
    kb.q[1] = *reinterpret_cast<const h8*>(kr1 + 16);
    s1 = WMMA_F16(qa0.v, kb.v, s1);
    kb.q[0] = *reinterpret_cast<const h8*>(kr1 + 32);
    kb.q[1] = *reinterpret_cast<const h8*>(kr1 + 48);
    s1 = WMMA_F16(qa1.v, kb.v, s1);

    // ---- streaming softmax (per-row reductions inside 16-lane halves) ----
#pragma unroll
    for (int r = 0; r < 8; r++) {
      float a0 = s0[r] * sc, a1 = s1[r] * sc;
      float cm = fmaxf(a0, a1);
      cm = fmaxf(cm, __shfl_xor(cm, 1));
      cm = fmaxf(cm, __shfl_xor(cm, 2));
      cm = fmaxf(cm, __shfl_xor(cm, 4));
      cm = fmaxf(cm, __shfl_xor(cm, 8));
      float mn = fmaxf(m[r], cm);
      float al = __expf(m[r] - mn);
      float p0 = __expf(a0 - mn);
      float p1 = __expf(a1 - mn);
      float rs = p0 + p1;
      rs += __shfl_xor(rs, 1); rs += __shfl_xor(rs, 2);
      rs += __shfl_xor(rs, 4); rs += __shfl_xor(rs, 8);
      l[r] = l[r] * al + rs;
      m[r] = mn;
      alpha[r] = al;
      plds[wave][r + hoff][lm] = (_Float16)p0;       // C-layout -> LDS tile
      plds[wave][r + hoff][16 + lm] = (_Float16)p1;
    }
#pragma unroll
    for (int r = 0; r < 8; r++) {
      o0[r] *= alpha[r]; o1[r] *= alpha[r]; o2[r] *= alpha[r]; o3[r] *= alpha[r];
    }

    // ---- P tile back as A-layout fragment ----
    AF pa;
    pa.q[0] = *reinterpret_cast<const h8*>(&plds[wave][lm][hoff]);
    pa.q[1] = *reinterpret_cast<const h8*>(&plds[wave][lm][16 + hoff]);

    // ---- O += P @ V, four 16-wide d tiles ----
    AF vb;
    const _Float16* vr0 = &vv[lm][hoff];
    vb.q[0] = *reinterpret_cast<const h8*>(vr0);
    vb.q[1] = *reinterpret_cast<const h8*>(vr0 + 16);
    o0 = WMMA_F16(pa.v, vb.v, o0);
    const _Float16* vr1 = &vv[16 + lm][hoff];
    vb.q[0] = *reinterpret_cast<const h8*>(vr1);
    vb.q[1] = *reinterpret_cast<const h8*>(vr1 + 16);
    o1 = WMMA_F16(pa.v, vb.v, o1);
    const _Float16* vr2 = &vv[32 + lm][hoff];
    vb.q[0] = *reinterpret_cast<const h8*>(vr2);
    vb.q[1] = *reinterpret_cast<const h8*>(vr2 + 16);
    o2 = WMMA_F16(pa.v, vb.v, o2);
    const _Float16* vr3 = &vv[48 + lm][hoff];
    vb.q[0] = *reinterpret_cast<const h8*>(vr3);
    vb.q[1] = *reinterpret_cast<const h8*>(vr3 + 16);
    o3 = WMMA_F16(pa.v, vb.v, o3);

    __syncthreads();  // all waves done reading this buffer before reissue
  }

  // ---- finalize: /= l, write as channels ch = d*4 + head ----
#pragma unroll
  for (int r = 0; r < 8; r++) {
    float inv = 1.f / l[r];
    int p = q0 + r + hoff;
    size_t base = (size_t)b * 256 * 4096 + p;
    oc[base + (size_t)((0 + lm) * 4 + head) * 4096]  = o0[r] * inv;
    oc[base + (size_t)((16 + lm) * 4 + head) * 4096] = o1[r] * inv;
    oc[base + (size_t)((32 + lm) * 4 + head) * 4096] = o2[r] * inv;
    oc[base + (size_t)((48 + lm) * 4 + head) * 4096] = o3[r] * inv;
  }
}

// ---------------------------------------------------------------------------
// Final fusion: bilinear up 64->128 of o-path and s-path, concat (192 ch),
// channel-last LayerNorm, write NCHW f32. One block per output pixel.
// ---------------------------------------------------------------------------
__global__ __launch_bounds__(192) void k_final(
    const float* __restrict__ oo, const float* __restrict__ sc,
    const float* __restrict__ g, const float* __restrict__ bt,
    float* __restrict__ out) {
  __shared__ float wsum[6], wsq[6];
  int c = threadIdx.x, p = blockIdx.x, b = blockIdx.y;
  int lane = c & 31, wid = c >> 5;
  int oy = p >> 7, ox = p & 127;
  const float scl = 63.f / 127.f;
  float py = oy * scl, px = ox * scl;
  int y0 = imin((int)py, 62), x0 = imin((int)px, 62);
  float wy = py - y0, wx = px - x0;
  const float* src = (c < 96 ? oo : sc) + ((size_t)b * 96 + (c < 96 ? c : c - 96)) * 4096;
  int bs = y0 * 64 + x0;
  float v00 = src[bs], v01 = src[bs + 1], v10 = src[bs + 64], v11 = src[bs + 65];
  float v = (1.f - wy) * ((1.f - wx) * v00 + wx * v01) + wy * ((1.f - wx) * v10 + wx * v11);

  float s = v, s2 = v * v;
  s += __shfl_xor(s, 1); s2 += __shfl_xor(s2, 1);
  s += __shfl_xor(s, 2); s2 += __shfl_xor(s2, 2);
  s += __shfl_xor(s, 4); s2 += __shfl_xor(s2, 4);
  s += __shfl_xor(s, 8); s2 += __shfl_xor(s2, 8);
  s += __shfl_xor(s, 16); s2 += __shfl_xor(s2, 16);
  if (lane == 0) { wsum[wid] = s; wsq[wid] = s2; }
  __syncthreads();
  if (threadIdx.x == 0) {
    float a = 0.f, a2 = 0.f;
    for (int i = 0; i < 6; i++) { a += wsum[i]; a2 += wsq[i]; }
    wsum[0] = a; wsq[0] = a2;
  }
  __syncthreads();
  float mu = wsum[0] * (1.f / 192.f);
  float var = wsq[0] * (1.f / 192.f) - mu * mu;
  float y = (v - mu) * rsqrtf(var + 1e-6f) * g[c] + bt[c];
  out[((size_t)b * 192 + c) * 16384 + p] = y;
}

// ---------------------------------------------------------------------------
// Host launcher
// ---------------------------------------------------------------------------
extern "C" void kernel_launch(void* const* d_in, const int* in_sizes, int n_in,
                              void* d_out, int out_size, void* d_ws, size_t ws_size,
                              hipStream_t stream) {
  (void)in_sizes; (void)n_in; (void)out_size; (void)ws_size;
  const float* xf    = (const float*)d_in[0];
  const float* xc    = (const float*)d_in[1];
  const float* pe_w  = (const float*)d_in[2];
  const float* pe_b  = (const float*)d_in[3];
  const float* pec_w = (const float*)d_in[4];
  const float* pec_b = (const float*)d_in[5];
  const float* q_dw  = (const float*)d_in[6];
  const float* q_pw  = (const float*)d_in[7];
  const float* k_dw  = (const float*)d_in[8];
  const float* k_pw  = (const float*)d_in[9];
  const float* v_dw  = (const float*)d_in[10];
  const float* v_pw  = (const float*)d_in[11];
  const float* o_dw  = (const float*)d_in[12];
  const float* o_pw  = (const float*)d_in[13];
  const float* s_dw  = (const float*)d_in[14];
  const float* s_pw  = (const float*)d_in[15];
  const float* gq    = (const float*)d_in[16];
  const float* bq    = (const float*)d_in[17];
  const float* gk    = (const float*)d_in[18];
  const float* bk    = (const float*)d_in[19];
  const float* g_out = (const float*)d_in[20];
  const float* b_out = (const float*)d_in[21];
  float* out = (float*)d_out;

  char* ws = (char*)d_ws;
  size_t off = 0;
  auto alloc = [&](size_t bytes) -> char* {
    char* p = ws + off;
    off = (off + bytes + 255) & ~(size_t)255;
    return p;
  };

  _Float16* wq16 = (_Float16*)alloc((size_t)256 * 192 * 2);
  _Float16* wk16 = (_Float16*)alloc((size_t)256 * 192 * 2);
  _Float16* wv16 = (_Float16*)alloc((size_t)256 * 96 * 2);
  _Float16* wo16 = (_Float16*)alloc((size_t)96 * 256 * 2);
  _Float16* ws16 = (_Float16*)alloc((size_t)96 * 192 * 2);
  _Float16* yqt  = (_Float16*)alloc((size_t)2 * 4096 * 192 * 2);
  _Float16* ykt  = (_Float16*)alloc((size_t)2 * 4096 * 192 * 2);
  _Float16* yst  = (_Float16*)alloc((size_t)2 * 4096 * 192 * 2);
  _Float16* yvt  = (_Float16*)alloc((size_t)2 * 16384 * 96 * 2);
  float*    qc   = (float*)alloc((size_t)2 * 256 * 4096 * 4);
  float*    kc   = (float*)alloc((size_t)2 * 256 * 4096 * 4);
  float*    scb  = (float*)alloc((size_t)2 * 96 * 4096 * 4);
  float*    vc   = (float*)alloc((size_t)2 * 256 * 16384 * 4);
  _Float16* Qh   = (_Float16*)alloc((size_t)2 * 4 * 4096 * 64 * 2);
  _Float16* Kh   = (_Float16*)alloc((size_t)2 * 4 * 4096 * 64 * 2);
  _Float16* Vt   = (_Float16*)alloc((size_t)2 * 4 * 64 * 4096 * 2);
  // lifetime-disjoint reuse:
  float*    ocb = qc;             // attention out [2,256,4096] f32 (qc dead after LN)
  _Float16* yot = (_Float16*)kc;  // dw-o out [2,4096,256] f16      (kc dead after LN)
  float*    oo  = vc;             // o pointwise out [2,96,4096] f32 (vc dead after resize)

  // 1. weight conversion
  k_f32_to_f16<<<(256 * 192 + 255) / 256, 256, 0, stream>>>(q_pw, wq16, 256 * 192);
  k_f32_to_f16<<<(256 * 192 + 255) / 256, 256, 0, stream>>>(k_pw, wk16, 256 * 192);
  k_f32_to_f16<<<(256 * 96 + 255) / 256, 256, 0, stream>>>(v_pw, wv16, 256 * 96);
  k_f32_to_f16<<<(96 * 256 + 255) / 256, 256, 0, stream>>>(o_pw, wo16, 96 * 256);
  k_f32_to_f16<<<(96 * 192 + 255) / 256, 256, 0, stream>>>(s_pw, ws16, 96 * 192);

  // 2. PE + depthwise
  k_dw_xc<<<dim3(4096, 2), 192, 0, stream>>>(xc, pec_w, pec_b, q_dw, k_dw, s_dw, yqt, ykt, yst);
  k_dw_xf<<<dim3(16384, 2), 96, 0, stream>>>(xf, pe_w, pe_b, v_dw, yvt);

  // 3. pointwise GEMMs (WMMA)
  k_gemm_pw<<<dim3(32, 16, 2), 256, 0, stream>>>(wq16, yqt, qc, 256, 192, 4096);
  k_gemm_pw<<<dim3(32, 16, 2), 256, 0, stream>>>(wk16, ykt, kc, 256, 192, 4096);
  k_gemm_pw<<<dim3(32, 6, 2), 256, 0, stream>>>(ws16, yst, scb, 96, 192, 4096);
  k_gemm_pw<<<dim3(128, 16, 2), 256, 0, stream>>>(wv16, yvt, vc, 256, 96, 16384);

  // 4. per-head LN (q,k) and V resize+transpose
  k_ln_heads<<<4096, 256, 0, stream>>>(qc, gq, bq, Qh);
  k_ln_heads<<<4096, 256, 0, stream>>>(kc, gk, bk, Kh);
  k_v_resize<<<(2 * 4 * 64 * 4096 + 255) / 256, 256, 0, stream>>>(vc, Vt);

  // 5. flash attention (WMMA + async LDS staging)
  k_attn<<<dim3(32, 8), 256, 0, stream>>>(Qh, Kh, Vt, ocb);

  // 6. output dsconv (WMMA pointwise)
  k_dw_o<<<dim3(4096, 2), 256, 0, stream>>>(ocb, o_dw, yot);
  k_gemm_pw<<<dim3(32, 6, 2), 256, 0, stream>>>(wo16, yot, oo, 96, 256, 4096);

  // 7. upsample + concat + LayerNorm
  k_final<<<dim3(16384, 2), 192, 0, stream>>>(oo, scb, g_out, b_out, out);
}
